// HomoEdgeDetector_65979287601566
// MI455X (gfx1250) — matrix operand order repaired and checked by reading, not compiled, not run on hardware
//
#include <hip/hip_runtime.h>
#include <hip/hip_bf16.h>

// ---------------------------------------------------------------------------
// Problem constants (match reference)
// ---------------------------------------------------------------------------
#define NNODES   50000
#define NEDGES   1600000
#define F_IN     256
#define F_HID    128
#define F_OUT    128
#define EL       (NEDGES + NNODES)      // edges + self loops
#define K_HOMO   960000                 // int(E * 0.6)
#define K_HET    640000
#define NBINS    2048

typedef __attribute__((ext_vector_type(16))) __bf16          v16bf;
typedef __attribute__((ext_vector_type(16))) unsigned short  v16us;
typedef __attribute__((ext_vector_type(8)))  float           v8f;

// ---------------------------------------------------------------------------
// helpers
// ---------------------------------------------------------------------------
__device__ __forceinline__ unsigned short f2bf(float f) {
  unsigned u = __float_as_uint(f);
  unsigned r = u + 0x7FFFu + ((u >> 16) & 1u);   // round-to-nearest-even
  return (unsigned short)(r >> 16);
}
// monotonic float<->uint key (order-preserving, handles negatives)
__device__ __forceinline__ unsigned fkey(float f) {
  unsigned u = __float_as_uint(f);
  return (u & 0x80000000u) ? ~u : (u | 0x80000000u);
}
__device__ __forceinline__ float funkey(unsigned k) {
  unsigned u = (k & 0x80000000u) ? (k & 0x7FFFFFFFu) : ~k;
  return __uint_as_float(u);
}

// ---------------------------------------------------------------------------
// 1) GEMM: H[N x 128] = X[N x K] @ W[K x 128]   (bf16 WMMA, f32 accumulate)
//    block = 256 threads = 8 waves; block owns 16 rows x 128 cols;
//    wave w owns the 16x16 tile at columns [16w,16w+16). K is a template
//    constant (256 / 128) so the K-loop fully unrolls: no loop SALU, and the
//    prefetch clamp folds to constants (straight-line global_prefetch_b8).
//    W tile is stored TRANSPOSED in LDS with a padded row stride (40 shorts =
//    80B, 16B-aligned rows) so each lane's 16-element B fragment is two
//    contiguous 16B chunks -> ds_load_b128 instead of 16x ds_load_u16.
// ---------------------------------------------------------------------------
#define AT_STRIDE 40   // shorts per A row  (16 rows)
#define BT_STRIDE 40   // shorts per Bt row (128 rows, Bt[col][k])

template <int K>
__global__ __launch_bounds__(256)
void gemm_wmma_bf16(const float* __restrict__ X, const float* __restrict__ W,
                    float* __restrict__ H) {
  __shared__ __align__(16) unsigned short As[16 * AT_STRIDE];
  __shared__ __align__(16) unsigned short Bt[128 * BT_STRIDE];

  const int tid  = threadIdx.x;
  const int wave = tid >> 5;
  const int lane = tid & 31;
  const int half = lane >> 4;     // 0: lanes 0-15, 1: lanes 16-31
  const int mn   = lane & 15;
  const int row0 = blockIdx.x * 16;   // N % 16 == 0 for N=50000
  const int col  = wave * 16 + mn;

  v8f acc = {};

#pragma unroll
  for (int k0 = 0; k0 < K; k0 += 32) {
    {  // straight-line next-tile prefetch (constants after unroll)
      const int kp = (k0 + 32 < K) ? (k0 + 32) : k0;
      __builtin_prefetch(&W[(size_t)(kp + (tid >> 7)) * 128 + (tid & 127)], 0, 1);
      __builtin_prefetch(&X[(size_t)(row0 + (tid & 15)) * K + kp], 0, 1);
    }
    // stage X tile: 512 elements, 2 per thread
#pragma unroll
    for (int i = 0; i < 2; ++i) {
      int e = tid + i * 256;
      int m = e >> 5, kk = e & 31;
      As[m * AT_STRIDE + kk] = f2bf(X[(size_t)(row0 + m) * K + k0 + kk]);
    }
    // stage W tile transposed: 4096 elements, 16 per thread
#pragma unroll
    for (int i = 0; i < 16; ++i) {
      int e = tid + i * 256;
      int kk = e >> 7, n = e & 127;
      Bt[n * BT_STRIDE + kk] = f2bf(W[(size_t)(k0 + kk) * 128 + n]);
    }
    __syncthreads();

    // pack fragments, contiguous k order for A and B: element i <-> k = 16*half + i
    v16us au, bu;
#pragma unroll
    for (int i = 0; i < 16; ++i) {
      int ka = 16 * half + i;
      au[i] = As[mn * AT_STRIDE + ka];              // 2x contiguous 16B chunks
      bu[i] = Bt[col * BT_STRIDE + 16 * half + i];  // 2x contiguous 16B chunks
    }
    v16bf a = __builtin_bit_cast(v16bf, au);
    v16bf b = __builtin_bit_cast(v16bf, bu);
    acc = __builtin_amdgcn_wmma_f32_16x16x32_bf16(
        /*neg_a=*/false, a, /*neg_b=*/false, b,
        /*c_mod=*/(short)0, acc, /*reuse_a=*/false, /*reuse_b=*/false);
    __syncthreads();
  }

  // C/D layout: VGPR r holds row (r + 8*half), col = mn
#pragma unroll
  for (int r = 0; r < 8; ++r) {
    int m = r + 8 * half;
    H[(size_t)(row0 + m) * 128 + col] = acc[r];
  }
}

// ---------------------------------------------------------------------------
// 2) per-node attention projections: hs[i] = h[i].a_s ; hd[i] = h[i].a_d
//    one wave per node (128 feats -> 4 per lane, shuffle reduce)
// ---------------------------------------------------------------------------
__global__ __launch_bounds__(256)
void attn_proj(const float* __restrict__ H, const float* __restrict__ a_s,
               const float* __restrict__ a_d, float* __restrict__ hs,
               float* __restrict__ hd, int N) {
  int gw   = (int)((blockIdx.x * (long long)blockDim.x + threadIdx.x) >> 5);
  int lane = threadIdx.x & 31;
  if (gw >= N) return;
  const float* row = H + (size_t)gw * 128;
  float s = 0.f, d = 0.f;
#pragma unroll
  for (int i = 0; i < 4; ++i) {
    float v = row[lane + 32 * i];
    s += v * a_s[lane + 32 * i];
    d += v * a_d[lane + 32 * i];
  }
#pragma unroll
  for (int off = 16; off > 0; off >>= 1) {
    s += __shfl_down(s, off, 32);
    d += __shfl_down(d, off, 32);
  }
  if (lane == 0) { hs[gw] = s; hd[gw] = d; }
}

// ---------------------------------------------------------------------------
// fill helper
// ---------------------------------------------------------------------------
__global__ void fill_u32(unsigned* __restrict__ p, unsigned v, size_t n) {
  size_t i = blockIdx.x * (size_t)blockDim.x + threadIdx.x;
  size_t stride = (size_t)gridDim.x * blockDim.x;
  for (; i < n; i += stride) p[i] = v;
}

// ---------------------------------------------------------------------------
// 3a) edge logits + segment max (monotonic-key atomic max)
// ---------------------------------------------------------------------------
__global__ void edge_logits_max(const int* __restrict__ srcA, const int* __restrict__ dstA,
                                const float* __restrict__ hs, const float* __restrict__ hd,
                                float* __restrict__ ebuf, unsigned* __restrict__ maxkey) {
  int i = blockIdx.x * blockDim.x + threadIdx.x;
  int stride = gridDim.x * blockDim.x;
  for (; i < EL; i += stride) {
    int s = (i < NEDGES) ? srcA[i] : (i - NEDGES);
    int d = (i < NEDGES) ? dstA[i] : (i - NEDGES);
    float e = hs[s] + hd[d];
    e = (e > 0.f) ? e : 0.2f * e;               // leaky_relu slope 0.2
    ebuf[i] = e;
    atomicMax(&maxkey[d], fkey(e));
  }
}

// 3b) exp + segment sum (in-place ebuf -> exp weights)
__global__ void edge_exp_sum(const int* __restrict__ dstA,
                             const unsigned* __restrict__ maxkey,
                             float* __restrict__ ebuf, float* __restrict__ den) {
  int i = blockIdx.x * blockDim.x + threadIdx.x;
  int stride = gridDim.x * blockDim.x;
  for (; i < EL; i += stride) {
    int d = (i < NEDGES) ? dstA[i] : (i - NEDGES);
    float ex = __expf(ebuf[i] - funkey(maxkey[d]));
    ebuf[i] = ex;
    atomicAdd(&den[d], ex);
  }
}

// 3c) weighted scatter-add aggregation: one wave per edge, 4 feats/lane.
//     Feature matrices (25.6MB) fit in the 192MB L2, so these random
//     gathers/atomics stay on-chip.
__global__ __launch_bounds__(256)
void aggregate(const int* __restrict__ srcA, const int* __restrict__ dstA,
               const float* __restrict__ ebuf, const float* __restrict__ den,
               const float* __restrict__ H, float* __restrict__ out) {
  long long gw = (blockIdx.x * (long long)blockDim.x + threadIdx.x) >> 5;
  int lane = threadIdx.x & 31;
  if (gw >= (long long)EL) return;
  int i = (int)gw;
  int s = (i < NEDGES) ? srcA[i] : (i - NEDGES);
  int d = (i < NEDGES) ? dstA[i] : (i - NEDGES);
  float alpha = ebuf[i] / (den[d] + 1e-16f);
  const float* hrow = H + (size_t)s * 128;
  float* orow = out + (size_t)d * 128;
#pragma unroll
  for (int j = 0; j < 4; ++j) {
    int f = lane + 32 * j;
    atomicAdd(&orow[f], hrow[f] * alpha);
  }
}

// ---------------------------------------------------------------------------
// 4) bias + leaky_relu epilogue (slope==1.0f -> identity)
// ---------------------------------------------------------------------------
__global__ void bias_act(const float* __restrict__ agg, const float* __restrict__ b,
                         float* __restrict__ out, float slope, size_t n) {
  size_t i = blockIdx.x * (size_t)blockDim.x + threadIdx.x;
  size_t stride = (size_t)gridDim.x * blockDim.x;
  for (; i < n; i += stride) {
    float v = agg[i] + b[i & 127];
    out[i] = (v > 0.f) ? v : v * slope;
  }
}

// ---------------------------------------------------------------------------
// 5) homophily keys: value = dot(x2[s], x2[d]); one wave per original edge
// ---------------------------------------------------------------------------
__global__ __launch_bounds__(256)
void edge_value(const int* __restrict__ srcA, const int* __restrict__ dstA,
                const float* __restrict__ X2, unsigned* __restrict__ keys) {
  long long gw = (blockIdx.x * (long long)blockDim.x + threadIdx.x) >> 5;
  int lane = threadIdx.x & 31;
  if (gw >= (long long)NEDGES) return;
  int i = (int)gw;
  const float* rs = X2 + (size_t)srcA[i] * 128;
  const float* rd = X2 + (size_t)dstA[i] * 128;
  float v = 0.f;
#pragma unroll
  for (int j = 0; j < 4; ++j) v += rs[lane + 32 * j] * rd[lane + 32 * j];
#pragma unroll
  for (int off = 16; off > 0; off >>= 1) v += __shfl_down(v, off, 32);
  if (lane == 0) keys[i] = fkey(v);
}

// 5b) histogram over top 11 key bits
__global__ void key_hist(const unsigned* __restrict__ keys, unsigned* __restrict__ hist) {
  int i = blockIdx.x * blockDim.x + threadIdx.x;
  int stride = gridDim.x * blockDim.x;
  for (; i < NEDGES; i += stride) atomicAdd(&hist[keys[i] >> 21], 1u);
}

// 5c) find threshold bin for rank K_HOMO (from the top)
__global__ void select_threshold(const unsigned* __restrict__ hist, int* __restrict__ ctrl) {
  if (threadIdx.x == 0 && blockIdx.x == 0) {
    long long cum = 0;
    int t = 0, needed = 0;
    for (int b = NBINS - 1; b >= 0; --b) {
      cum += hist[b];
      if (cum >= (long long)K_HOMO) {
        t = b;
        needed = (int)(K_HOMO - (cum - hist[b]));
        break;
      }
    }
    ctrl[0] = t;       // threshold bin
    ctrl[1] = needed;  // how many to take from bin t
    // ctrl[2] = bin-taken counter, ctrl[3] = homo pos, ctrl[4] = het pos (pre-zeroed)
  }
}

// 5d) partition edges into homo/het int64 outputs
__global__ void partition_edges(const int* __restrict__ srcA, const int* __restrict__ dstA,
                                const unsigned* __restrict__ keys, int* __restrict__ ctrl,
                                long long* __restrict__ homo_s, long long* __restrict__ homo_d,
                                long long* __restrict__ het_s,  long long* __restrict__ het_d) {
  int i = blockIdx.x * blockDim.x + threadIdx.x;
  int stride = gridDim.x * blockDim.x;
  int t = ctrl[0], needed = ctrl[1];
  for (; i < NEDGES; i += stride) {
    unsigned k = keys[i];
    int bin = (int)(k >> 21);
    bool homo;
    if (bin > t)       homo = true;
    else if (bin < t)  homo = false;
    else               homo = (atomicAdd(&ctrl[2], 1) < needed);
    long long sv = srcA[i], dv = dstA[i];
    if (homo) {
      int p = atomicAdd(&ctrl[3], 1);
      homo_s[p] = sv; homo_d[p] = dv;
    } else {
      int p = atomicAdd(&ctrl[4], 1);
      het_s[p] = sv; het_d[p] = dv;
    }
  }
}

// 5e) copy x2 to tail of d_out
__global__ void copy_f32(const float* __restrict__ src, float* __restrict__ dst, size_t n) {
  size_t i = blockIdx.x * (size_t)blockDim.x + threadIdx.x;
  size_t stride = (size_t)gridDim.x * blockDim.x;
  for (; i < n; i += stride) dst[i] = src[i];
}

// ---------------------------------------------------------------------------
// launcher
// ---------------------------------------------------------------------------
extern "C" void kernel_launch(void* const* d_in, const int* in_sizes, int n_in,
                              void* d_out, int out_size, void* d_ws, size_t ws_size,
                              hipStream_t stream) {
  (void)in_sizes; (void)n_in; (void)out_size; (void)ws_size;

  const float* x      = (const float*)d_in[0];
  const int*   ei     = (const int*)d_in[1];       // [2, E]: src row then dst row
  const float* W1     = (const float*)d_in[2];
  const float* a_src1 = (const float*)d_in[3];
  const float* a_dst1 = (const float*)d_in[4];
  const float* b1     = (const float*)d_in[5];
  const float* W2     = (const float*)d_in[6];
  const float* a_src2 = (const float*)d_in[7];
  const float* a_dst2 = (const float*)d_in[8];
  const float* b2     = (const float*)d_in[9];

  const int* srcA = ei;
  const int* dstA = ei + NEDGES;

  // --- carve workspace ---
  char* w = (char*)d_ws;
  auto carve = [&](size_t bytes) -> char* {
    char* p = w;
    w += (bytes + 255) & ~(size_t)255;
    return p;
  };
  float*    bufH   = (float*)carve((size_t)NNODES * 128 * 4);  // h (pre-agg features)
  float*    bufAgg = (float*)carve((size_t)NNODES * 128 * 4);  // aggregated output
  float*    bufX   = (float*)carve((size_t)NNODES * 128 * 4);  // x1, then x2
  float*    hs     = (float*)carve((size_t)NNODES * 4);
  float*    hd     = (float*)carve((size_t)NNODES * 4);
  unsigned* maxkey = (unsigned*)carve((size_t)NNODES * 4);
  float*    den    = (float*)carve((size_t)NNODES * 4);
  float*    ebuf   = (float*)carve((size_t)EL * 4);            // logits -> exp -> keys
  unsigned* hist   = (unsigned*)carve((size_t)NBINS * 4);
  int*      ctrl   = (int*)carve(64);

  // --- d_out layout: [homo_src|homo_dst|het_src|het_dst] int64, then x2 f32 ---
  long long* out_ll  = (long long*)d_out;
  long long* homo_s  = out_ll;
  long long* homo_d  = out_ll + K_HOMO;
  long long* het_s   = out_ll + 2 * (size_t)K_HOMO;
  long long* het_d   = out_ll + 2 * (size_t)K_HOMO + K_HET;
  float*     out_x2  = (float*)(out_ll + 2 * (size_t)K_HOMO + 2 * (size_t)K_HET);

  const int EB   = 2048;                                    // grid-stride edge blocks
  const int AGGB = (int)(((long long)EL * 32 + 255) / 256); // wave-per-edge
  const int VALB = (int)(((long long)NEDGES * 32 + 255) / 256);
  const int ATTB = (NNODES * 32 + 255) / 256;

  for (int layer = 0; layer < 2; ++layer) {
    const float* Xin = (layer == 0) ? x : bufX;
    const float* W   = (layer == 0) ? W1 : W2;
    const float* as_ = (layer == 0) ? a_src1 : a_src2;
    const float* ad_ = (layer == 0) ? a_dst1 : a_dst2;
    const float* bb  = (layer == 0) ? b1 : b2;
    const float  sl  = (layer == 0) ? 0.01f : 1.0f;

    if (layer == 0)
      gemm_wmma_bf16<F_IN ><<<NNODES / 16, 256, 0, stream>>>(Xin, W, bufH);
    else
      gemm_wmma_bf16<F_HID><<<NNODES / 16, 256, 0, stream>>>(Xin, W, bufH);

    attn_proj<<<ATTB, 256, 0, stream>>>(bufH, as_, ad_, hs, hd, NNODES);

    fill_u32<<<256, 256, 0, stream>>>(maxkey, 0u, NNODES);
    fill_u32<<<256, 256, 0, stream>>>((unsigned*)den, 0u, NNODES);
    fill_u32<<<1024, 256, 0, stream>>>((unsigned*)bufAgg, 0u, (size_t)NNODES * 128);

    edge_logits_max<<<EB, 256, 0, stream>>>(srcA, dstA, hs, hd, ebuf, maxkey);
    edge_exp_sum<<<EB, 256, 0, stream>>>(dstA, maxkey, ebuf, den);
    aggregate<<<AGGB, 256, 0, stream>>>(srcA, dstA, ebuf, den, bufH, bufAgg);
    bias_act<<<1024, 256, 0, stream>>>(bufAgg, bb, bufX, sl, (size_t)NNODES * 128);
  }

  // homophily scoring + partition
  unsigned* keys = (unsigned*)ebuf;  // reuse
  edge_value<<<VALB, 256, 0, stream>>>(srcA, dstA, bufX, keys);
  fill_u32<<<8, 256, 0, stream>>>(hist, 0u, NBINS);
  fill_u32<<<1, 64, 0, stream>>>((unsigned*)ctrl, 0u, 16);
  key_hist<<<EB, 256, 0, stream>>>(keys, hist);
  select_threshold<<<1, 1, 0, stream>>>(hist, ctrl);
  partition_edges<<<EB, 256, 0, stream>>>(srcA, dstA, keys, ctrl,
                                          homo_s, homo_d, het_s, het_d);
  copy_f32<<<1024, 256, 0, stream>>>(bufX, out_x2, (size_t)NNODES * 128);
}
